// HypernymVisual_acc_28226525069938
// MI455X (gfx1250) — matrix operand order repaired and checked by reading, not compiled, not run on hardware
//
#include <hip/hip_runtime.h>

typedef __attribute__((ext_vector_type(16))) __bf16 v16bf;
typedef __attribute__((ext_vector_type(8)))  float  v8f;

constexpr int kVF  = 4096;     // VF_D
constexpr int kHID = 2048;     // HID_D
constexpr int kEMB = 512;      // EMB_D
constexpr int kB   = 256;      // batch
constexpr int kNN  = 512;      // negatives
constexpr int kCols = kNN + 1; // 513 output columns

union BFVec {
    v16bf v;
    unsigned short u[16];
    unsigned w[8];
};

__device__ __forceinline__ unsigned short bf16_rne(float f) {
    unsigned u = __float_as_uint(f);
    unsigned r = 0x7FFFu + ((u >> 16) & 1u);
    return (unsigned short)((u + r) >> 16);
}

// Pack two f32 -> one dword of two bf16 {hi,lo} using the CDNA5 VOP3P
// mixed-precision converters (V_FMA_MIXLO_BF16 / V_FMA_MIXHI_BF16).
__device__ __forceinline__ unsigned pack_bf16(float lo, float hi) {
    unsigned d;
    asm("v_fma_mixlo_bf16 %0, %1, 1.0, 0\n\t"
        "v_fma_mixhi_bf16 %0, %2, 1.0, 0"
        : "=v"(d)
        : "v"(lo), "v"(hi));
    return d;
}

__device__ __forceinline__ void cvt_f4(unsigned* w, float4 f) {
    w[0] = pack_bf16(f.x, f.y);
    w[1] = pack_bf16(f.z, f.w);
}

// Raw (unconverted) data for one 32-wide k-step of a 32(M)x64(N) wave tile.
// Only one of af/ab is used per template instantiation (dead one is DCE'd).
struct Buf {
    float4 af[2][4];  // A rows, f32 source
    uint4  ab[2][2];  // A rows, bf16 source
    float4 b[4][4];   // W rows, f32 source
};

// Partial GEMM: Cpart[kchunk][M x N] = A[M x kc] * W[N x kc]^T over this
// block's K-chunk (no bias). bf16 WMMA, f32 accumulate.
// Each wave computes a 32(M) x 64(N) macro-tile: 2 A-frags x 4 B-frags.
// Software-pipelined one k-step deep: loads for step i+1 are issued before
// the converts/WMMAs of step i, keeping ~24 b128 loads in flight per wave.
template <bool A_BF16>
__global__ __launch_bounds__(128)
void gemm_nt_wmma(const void* __restrict__ Ap, const float* __restrict__ W,
                  float* __restrict__ Cpart, int N, int K, int Kc,
                  int MN, int tilesN, int totalTiles)
{
    const int wave = threadIdx.x >> 5;
    const int lane = threadIdx.x & 31;
    const int hl   = lane >> 4;   // 16-lane half
    const int r    = lane & 15;
    const int tile = blockIdx.x * 4 + wave;
    if (tile >= totalTiles) return;
    const int m0 = (tile / tilesN) * 32;
    const int n0 = (tile % tilesN) * 64;
    const int kbeg = blockIdx.y * Kc;

    // Per-lane row base pointers (K-offset added per step).
    const float*          aRowF[2];
    const unsigned short* aRowB[2];
    const float*          wRow[4];
    #pragma unroll
    for (int mm = 0; mm < 2; ++mm) {
        const int row = m0 + 16 * mm + r;
        aRowF[mm] = (const float*)Ap + (size_t)row * K + 8 * hl;
        aRowB[mm] = (const unsigned short*)Ap + (size_t)row * K + 8 * hl;
    }
    #pragma unroll
    for (int s = 0; s < 4; ++s)
        wRow[s] = W + (size_t)(n0 + 16 * s + r) * K + 8 * hl;

    v8f acc[2][4] = {};

    auto load_step = [&](Buf& bf, int kk) {
        #pragma unroll
        for (int mm = 0; mm < 2; ++mm) {
            if (A_BF16) {
                const unsigned short* ab = aRowB[mm] + kk;
                bf.ab[mm][0] = *(const uint4*)(ab);
                bf.ab[mm][1] = *(const uint4*)(ab + 16);
            } else {
                const float* af = aRowF[mm] + kk;
                bf.af[mm][0] = *(const float4*)(af);
                bf.af[mm][1] = *(const float4*)(af + 4);
                bf.af[mm][2] = *(const float4*)(af + 16);
                bf.af[mm][3] = *(const float4*)(af + 20);
            }
        }
        #pragma unroll
        for (int s = 0; s < 4; ++s) {
            const float* wr = wRow[s] + kk;
            bf.b[s][0] = *(const float4*)(wr);
            bf.b[s][1] = *(const float4*)(wr + 4);
            bf.b[s][2] = *(const float4*)(wr + 16);
            bf.b[s][3] = *(const float4*)(wr + 20);
        }
    };

    auto compute_step = [&](const Buf& bf) {
        // A fragments (16x32 bf16): half 0 holds K +0..7 / +16..23,
        // half 1 holds K +8..15 / +24..31.
        BFVec a[2];
        #pragma unroll
        for (int mm = 0; mm < 2; ++mm) {
            if (A_BF16) {
                a[mm].w[0] = bf.ab[mm][0].x; a[mm].w[1] = bf.ab[mm][0].y;
                a[mm].w[2] = bf.ab[mm][0].z; a[mm].w[3] = bf.ab[mm][0].w;
                a[mm].w[4] = bf.ab[mm][1].x; a[mm].w[5] = bf.ab[mm][1].y;
                a[mm].w[6] = bf.ab[mm][1].z; a[mm].w[7] = bf.ab[mm][1].w;
            } else {
                cvt_f4(a[mm].w + 0, bf.af[mm][0]); cvt_f4(a[mm].w + 2, bf.af[mm][1]);
                cvt_f4(a[mm].w + 4, bf.af[mm][2]); cvt_f4(a[mm].w + 6, bf.af[mm][3]);
            }
        }
        // B fragments (32x16 bf16): lanes 0-15 K+0..7 / +16..23, lanes 16-31 K+8..15 / +24..31.
        #pragma unroll
        for (int s = 0; s < 4; ++s) {
            BFVec b;
            cvt_f4(b.w + 0, bf.b[s][0]); cvt_f4(b.w + 2, bf.b[s][1]);
            cvt_f4(b.w + 4, bf.b[s][2]); cvt_f4(b.w + 6, bf.b[s][3]);
            acc[0][s] = __builtin_amdgcn_wmma_f32_16x16x32_bf16(
                false, a[0].v, false, b.v, (short)0, acc[0][s], false, false);
            acc[1][s] = __builtin_amdgcn_wmma_f32_16x16x32_bf16(
                false, a[1].v, false, b.v, (short)0, acc[1][s], false, false);
        }
    };

    // nsteps is even for every launch configuration used below.
    const int nsteps = Kc >> 5;
    Buf b0, b1;
    load_step(b0, kbeg);
    int kk = kbeg;
    for (int it = 0; it < nsteps; it += 2) {
        load_step(b1, kk + 32);
        compute_step(b0);
        if (it + 2 < nsteps) load_step(b0, kk + 64);
        compute_step(b1);
        kk += 64;
    }

    // Epilogue: C/D layout VGPR i -> M = i + 8*hl, N = n0+16s+r.
    float* out = Cpart + (size_t)blockIdx.y * MN;
    #pragma unroll
    for (int s = 0; s < 4; ++s) {
        const int n = n0 + 16 * s + r;
        #pragma unroll
        for (int mm = 0; mm < 2; ++mm) {
            #pragma unroll
            for (int i = 0; i < 8; ++i) {
                const int m = m0 + 16 * mm + i + 8 * hl;
                out[(size_t)m * N + n] = acc[mm][s][i];
            }
        }
    }
}

// out[i] = sum_c part[c][i] + bias[i % N]   (N is a power of two)
template <bool OUT_BF16>
__global__ __launch_bounds__(256)
void reduce_bias(const float* __restrict__ part, const float* __restrict__ bias,
                 void* __restrict__ out, int MN, int Nmask, int chunks)
{
    const int i = blockIdx.x * 256 + threadIdx.x;
    if (i >= MN) return;
    float s = part[i];
    for (int c = 1; c < chunks; ++c) s += part[(size_t)c * MN + i];
    s += bias[i & Nmask];
    if (OUT_BF16)
        ((unsigned short*)out)[i] = bf16_rne(s);
    else
        ((float*)out)[i] = s;
}

// out[b][0]   = -||relu(p_lfs[b] - emb[b])||_2
// out[b][1+n] = -||relu(n_lfs[n] - emb[b])||_2
// One wave32 per output element; lane-strided coalesced reads + shfl reduction.
__global__ __launch_bounds__(256)
void scores_kernel(const float* __restrict__ emb, const float* __restrict__ p_lfs,
                   const float* __restrict__ n_lfs, float* __restrict__ out)
{
    const int wave = threadIdx.x >> 5;
    const int lane = threadIdx.x & 31;
    const int gw   = blockIdx.x * 8 + wave;
    if (gw >= kB * kCols) return;
    const int b   = gw / kCols;
    const int col = gw % kCols;

    const float* e = emb + (size_t)b * kEMB;
    const float* t = (col == 0) ? (p_lfs + (size_t)b * kEMB)
                                : (n_lfs + (size_t)(col - 1) * kEMB);
    float s = 0.0f;
    #pragma unroll 4
    for (int d = lane; d < kEMB; d += 32) {
        float diff = t[d] - e[d];
        diff = fmaxf(diff, 0.0f);
        s = fmaf(diff, diff, s);
    }
    #pragma unroll
    for (int off = 16; off > 0; off >>= 1)
        s += __shfl_down(s, off, 32);
    if (lane == 0)
        out[(size_t)b * kCols + col] = -sqrtf(s);
}

extern "C" void kernel_launch(void* const* d_in, const int* in_sizes, int n_in,
                              void* d_out, int out_size, void* d_ws, size_t ws_size,
                              hipStream_t stream) {
    (void)in_sizes; (void)n_in; (void)out_size;

    const float* vfs      = (const float*)d_in[0];  // [256, 4096]
    const float* p_lfs    = (const float*)d_in[1];  // [256, 512]
    const float* n_lfs    = (const float*)d_in[2];  // [512, 512]
    const float* W_hidden = (const float*)d_in[3];  // [2048, 4096]
    const float* b_hidden = (const float*)d_in[4];  // [2048]
    const float* W_embed  = (const float*)d_in[5];  // [512, 2048]
    const float* b_embed  = (const float*)d_in[6];  // [512]

    const int MN1 = kB * kHID;  // 524288
    const int MN2 = kB * kEMB;  // 131072

    // Split-K factor: largest of {8,4,1} whose partial buffers fit d_ws.
    auto need = [&](int ks) {
        return (size_t)ks * MN1 * 4 + (size_t)MN1 * 2 +
               (size_t)ks * MN2 * 4 + (size_t)MN2 * 4;
    };
    const int KS = (ws_size >= need(8)) ? 8 : (ws_size >= need(4)) ? 4 : 1;

    char* p = (char*)d_ws;
    float*          part1  = (float*)p;          p += (size_t)KS * MN1 * 4;
    unsigned short* hidden = (unsigned short*)p; p += (size_t)MN1 * 2;
    float*          part2  = (float*)p;          p += (size_t)KS * MN2 * 4;
    float*          emb    = (float*)p;

    // GEMM1 partials: tiles (256/32)x(2048/64) = 256, 4 waves/block -> 64 blocks x KS chunks
    gemm_nt_wmma<false><<<dim3(64, KS), 128, 0, stream>>>(
        vfs, W_hidden, part1, kHID, kVF, kVF / KS, MN1, kHID / 64, (kB / 32) * (kHID / 64));

    // hidden = sum(part1) + b_hidden -> bf16
    reduce_bias<true><<<MN1 / 256, 256, 0, stream>>>(
        part1, b_hidden, hidden, MN1, kHID - 1, KS);

    // GEMM2 partials: tiles (256/32)x(512/64) = 64 -> 16 blocks x KS chunks
    gemm_nt_wmma<true><<<dim3(16, KS), 128, 0, stream>>>(
        hidden, W_embed, part2, kEMB, kHID, kHID / KS, MN2, kEMB / 64, (kB / 32) * (kEMB / 64));

    // emb = sum(part2) + b_embed -> f32
    reduce_bias<false><<<MN2 / 256, 256, 0, stream>>>(
        part2, b_embed, emb, MN2, kEMB - 1, KS);

    // Scores: 256*513 = 131328 wave-outputs, 8 waves/block -> 16416 blocks
    scores_kernel<<<16416, 256, 0, stream>>>(emb, p_lfs, n_lfs, (float*)d_out);
}
